// InverseReceptiveField_48808008351837
// MI455X (gfx1250) — matrix operand order repaired and checked by reading, not compiled, not run on hardware
//
#include <hip/hip_runtime.h>
#include <hip/hip_bf16.h>

// Problem constants (from reference setup_inputs):
//   h=8, n=8, d=512, c=64, c2=32, i=3, kk=3
// Outputs flat-concat: T(8*64*3*3=4608) | M(8*8*32=2048) | P(8*8*64*9=36864) | w(36864)

typedef __attribute__((ext_vector_type(16))) _Float16 v16h;
typedef __attribute__((ext_vector_type(8)))  float    v8f;
typedef int v4i __attribute__((vector_size(16)));

#define BN_EPS 1e-5f

#define AS1 __attribute__((address_space(1)))
#define AS3 __attribute__((address_space(3)))

#define SW_STRIDE 580   // 576 + pad; 580*4 B is 16B-aligned, breaks bank-aligned stride

// ---------------------------------------------------------------------------
// Kernel 1: per-head Linear (WMMA f16->f32) + BatchNorm(n) + ReLU -> M, P
// grid = 8 (one block per head), block = 32 (exactly one wave32)
// ---------------------------------------------------------------------------
__global__ void __launch_bounds__(32)
embed_kernel(const float* __restrict__ X,      // (8, 8, 512)
             const float* __restrict__ lin_w,  // (8, 32, 512)
             const float* __restrict__ lin_b,  // (8, 32)
             const float* __restrict__ gamma,  // (8, 32)
             const float* __restrict__ beta,   // (8, 32)
             const float* __restrict__ px,     // (32, 1, 3)
             const float* __restrict__ py,     // (32, 3, 1)
             float* __restrict__ M_out,        // (8, 8, 32)
             float* __restrict__ P_out)        // (8, 8, 64, 3, 3)
{
    const int h    = blockIdx.x;
    const int lane = threadIdx.x;
    const int mrow = lane & 15;
    const bool hi  = lane >= 16;

    const float* Xh = X     + (size_t)h * 8  * 512;
    const float* Wh = lin_w + (size_t)h * 32 * 512;

    v8f acc0 = {}, acc1 = {};

    // y[h] = X[h] (8x512, padded to 16 rows) @ lin_w[h]^T (512x32)
    for (int k0 = 0; k0 < 512; k0 += 32) {
        // A fragment (16x32 f16): lane<16 holds K = k0+{0..15},
        // lane>=16 holds K = k0+{16..31}; M = lane%16 (rows >=8 are zero pad)
        v16h a = {};
        if (mrow < 8) {
            const float* ap0 = Xh + mrow * 512 + k0 + (hi ? 16 : 0);
            const float* ap1 = Xh + mrow * 512 + k0 + (hi ? 24 : 8);
#pragma unroll
            for (int t = 0; t < 8; ++t) {
                a[t]     = (_Float16)ap0[t];
                a[8 + t] = (_Float16)ap1[t];
            }
        }
        // B fragments (32x16 f16): column N = lane%16, K-half by lane-half.
        // B[k][n] = lin_w[h][n][k] -> contiguous in k.
#pragma unroll
        for (int nt = 0; nt < 2; ++nt) {
            const float* bp = Wh + (nt * 16 + mrow) * 512 + k0 + (hi ? 16 : 0);
            v16h b;
#pragma unroll
            for (int t = 0; t < 16; ++t) b[t] = (_Float16)bp[t];
            if (nt == 0)
                acc0 = __builtin_amdgcn_wmma_f32_16x16x32_f16(
                    false, a, false, b, (short)0, acc0, false, false);
            else
                acc1 = __builtin_amdgcn_wmma_f32_16x16x32_f16(
                    false, a, false, b, (short)0, acc1, false, false);
        }
    }

    // C/D layout: lane<16, VGPR r -> y[row=r][col=lane]; all 8 valid batch rows
    // of a column live in ONE lane -> BatchNorm over n is lane-local.
    if (!hi) {
#pragma unroll
        for (int nt = 0; nt < 2; ++nt) {
            const v8f accv = (nt == 0) ? acc0 : acc1;
            const int col  = nt * 16 + mrow;
            const float bias = lin_b[h * 32 + col];

            float v[8];
            float mean = 0.f;
#pragma unroll
            for (int r = 0; r < 8; ++r) { v[r] = accv[r] + bias; mean += v[r]; }
            mean *= 0.125f;
            float var = 0.f;
#pragma unroll
            for (int r = 0; r < 8; ++r) { float dv = v[r] - mean; var += dv * dv; }
            var *= 0.125f;  // biased, as BatchNorm training stats

            const float sc = gamma[h * 32 + col] * rsqrtf(var + BN_EPS);
            const float bb = beta[h * 32 + col];

#pragma unroll
            for (int r = 0; r < 8; ++r) {  // r == batch index n
                float m = (v[r] - mean) * sc + bb;
                m = m > 0.f ? m : 0.f;     // ReLU
                M_out[(size_t)(r * 8 + h) * 32 + col] = m;

                // P: channels [0,32) = m + px (varies over jj); [32,64) = m + py (varies over ii)
                const size_t bx = ((size_t)(r * 8 + h) * 64 + col) * 9;
                const size_t by = ((size_t)(r * 8 + h) * 64 + 32 + col) * 9;
#pragma unroll
                for (int ii = 0; ii < 3; ++ii)
#pragma unroll
                    for (int jj = 0; jj < 3; ++jj) {
                        P_out[bx + ii * 3 + jj] = m + px[col * 3 + jj];
                        P_out[by + ii * 3 + jj] = m + py[col * 3 + ii];
                    }
            }
        }
    }
}

// ---------------------------------------------------------------------------
// Kernel 2: per-head grouped 1x1 key/value convs as WMMA GEMMs.
// Per head: A = P rows (n*9 = 72, padded to 80) x K=64 ; B = kw^T / vw^T (64x64).
// grid = 8 heads * 5 M-tiles = 40 blocks, block = 32 (one wave32)
// ---------------------------------------------------------------------------
__global__ void __launch_bounds__(32)
kpvp_kernel(const float* __restrict__ P,       // (8, 8, 64, 3, 3)
            const float* __restrict__ kw,      // (8, 64, 64)
            const float* __restrict__ kb,      // (8, 64)
            const float* __restrict__ vw,      // (8, 64, 64)
            const float* __restrict__ vb,      // (8, 64)
            float* __restrict__ kp_ws,         // (8, 8, 64, 9)
            float* __restrict__ vp_ws)         // (8, 8, 64, 9)
{
    const int h    = blockIdx.x / 5;
    const int mt   = blockIdx.x % 5;
    const int lane = threadIdx.x;
    const int mrow = lane & 15;
    const bool hi  = lane >= 16;

    v8f ak[4] = {{}, {}, {}, {}};
    v8f av[4] = {{}, {}, {}, {}};

    // A row for this lane: row = n*9 + p  (valid rows: 0..71)
    const int  arow   = mt * 16 + mrow;
    const bool avalid = arow < 72;
    const int  an = arow / 9, apos = arow % 9;
    const float* Pbase = P + ((size_t)(an * 8 + h) * 64) * 9 + apos;  // +c*9 per channel

#pragma unroll
    for (int k0 = 0; k0 < 64; k0 += 32) {
        v16h a = {};
        if (avalid) {
            int cb0 = k0 + (hi ? 16 : 0);
            int cb1 = k0 + (hi ? 24 : 8);
#pragma unroll
            for (int t = 0; t < 8; ++t) {
                a[t]     = (_Float16)Pbase[(size_t)(cb0 + t) * 9];
                a[8 + t] = (_Float16)Pbase[(size_t)(cb1 + t) * 9];
            }
        }
        const int crange = k0 + (hi ? 16 : 0);
#pragma unroll
        for (int nt = 0; nt < 4; ++nt) {
            const int dcol = nt * 16 + mrow;
            const float* kwp = kw + (size_t)h * 4096 + dcol * 64 + crange;
            const float* vwp = vw + (size_t)h * 4096 + dcol * 64 + crange;
            v16h bk, bv;
#pragma unroll
            for (int t = 0; t < 16; ++t) {
                bk[t] = (_Float16)kwp[t];
                bv[t] = (_Float16)vwp[t];
            }
            ak[nt] = __builtin_amdgcn_wmma_f32_16x16x32_f16(
                false, a, false, bk, (short)0, ak[nt], false, false);
            av[nt] = __builtin_amdgcn_wmma_f32_16x16x32_f16(
                false, a, false, bv, (short)0, av[nt], false, false);
        }
    }

    // Scatter accumulators: row = mt*16 + r (+8 for lane-high half), col = nt*16 + lane%16
#pragma unroll
    for (int nt = 0; nt < 4; ++nt) {
        const int dcol = nt * 16 + mrow;
        const float kbv = kb[h * 64 + dcol];
        const float vbv = vb[h * 64 + dcol];
#pragma unroll
        for (int r = 0; r < 8; ++r) {
            const int row = mt * 16 + r + (hi ? 8 : 0);
            if (row < 72) {
                const int nn = row / 9, pp = row % 9;
                const size_t idx = ((size_t)(nn * 8 + h) * 64 + dcol) * 9 + pp;
                kp_ws[idx] = ak[nt][r] + kbv;
                vp_ws[idx] = av[nt][r] + vbv;
            }
        }
    }
}

// ---------------------------------------------------------------------------
// Kernel 3: 3x3 SAME conv on kp, scale, softmax over heads, weighted vp sum.
// grid = 8 blocks (one per n), block = 576 threads (64 d-channels x 9 positions).
// sw (147 KB) and kp[n] (18 KB) are staged into LDS once per block via the
// CDNA5 async global->LDS copy path (GLOBAL_LOAD_ASYNC_TO_LDS_B128 +
// s_wait_asynccnt), with a plain-LDS-copy fallback if the builtins are absent.
// ---------------------------------------------------------------------------
__global__ void __launch_bounds__(576)
attn_kernel(const float* __restrict__ kp,     // (8, 8, 64, 9)
            const float* __restrict__ vp,     // (8, 8, 64, 9)
            const float* __restrict__ sw,     // (64, 64, 3, 3)
            const float* __restrict__ sb,     // (64,)
            float* __restrict__ T_out,        // (8, 64, 3, 3)
            float* __restrict__ W_out)        // (8, 8, 64, 3, 3)
{
    extern __shared__ float smem[];
    float* s_sw = smem;                      // 64 rows x SW_STRIDE (padded)
    float* s_kp = smem + 64 * SW_STRIDE;     // 8*64*9 = 4608 floats

    const int n   = blockIdx.x;
    const int tid = threadIdx.x;             // 0..575
    const float* kpn = kp + (size_t)n * 4608;

    // ---- cooperative stage: sw rows (576 f32 each -> padded stride) + kp[n] ----
#if __has_builtin(__builtin_amdgcn_global_load_async_to_lds_b128)
#pragma unroll
    for (int it = 0; it < 16; ++it) {
        const int v  = it * 576 + tid;       // vec4 id, 0..9215
        const int dd = v / 144;
        const int w  = (v % 144) * 4;        // float offset within row, 16B aligned
        __builtin_amdgcn_global_load_async_to_lds_b128(
            (AS1 v4i*)(sw + (size_t)dd * 576 + w),
            (AS3 v4i*)(s_sw + dd * SW_STRIDE + w), 0, 0);
    }
#pragma unroll
    for (int it = 0; it < 2; ++it) {
        const int w = (it * 576 + tid) * 4;  // 0..4604, 16B aligned
        __builtin_amdgcn_global_load_async_to_lds_b128(
            (AS1 v4i*)(kpn + w),
            (AS3 v4i*)(s_kp + w), 0, 0);
    }
#if __has_builtin(__builtin_amdgcn_s_wait_asynccnt)
    __builtin_amdgcn_s_wait_asynccnt(0);
#else
    asm volatile("s_wait_asynccnt 0" ::: "memory");
#endif
#else   // fallback: plain vector loads + LDS stores
#pragma unroll
    for (int it = 0; it < 16; ++it) {
        const int v  = it * 576 + tid;
        const int dd = v / 144;
        const int w  = (v % 144) * 4;
        *(float4*)(s_sw + dd * SW_STRIDE + w) =
            *(const float4*)(sw + (size_t)dd * 576 + w);
    }
#pragma unroll
    for (int it = 0; it < 2; ++it) {
        const int w = (it * 576 + tid) * 4;
        *(float4*)(s_kp + w) = *(const float4*)(kpn + w);
    }
#endif
    __syncthreads();

    const int d = tid & 63;                  // output channel
    const int p = tid >> 6;                  // spatial position 0..8
    const int y = p / 3, x = p % 3;
    const float invs = 1.0f / 24.0f;         // 1/sqrt(c*kk*kk) = 1/sqrt(576)
    const float* swd = s_sw + d * SW_STRIDE;

    float s[8];
#pragma unroll
    for (int h = 0; h < 8; ++h) {
        float a = 0.f;
        const float* kph = s_kp + h * 576;   // (64 c x 9 pos)
#pragma unroll
        for (int dy = 0; dy < 3; ++dy) {
            const int yy = y + dy - 1;
            if (yy < 0 || yy > 2) continue;
#pragma unroll
            for (int dx = 0; dx < 3; ++dx) {
                const int xx = x + dx - 1;
                if (xx < 0 || xx > 2) continue;
                const int pp = yy * 3 + xx;
                const int o  = dy * 3 + dx;
                for (int c = 0; c < 64; ++c)
                    a = fmaf(kph[c * 9 + pp], swd[c * 9 + o], a);
            }
        }
        s[h] = (a + sb[d]) * invs;
    }

    // softmax over heads
    float mx = s[0];
#pragma unroll
    for (int h = 1; h < 8; ++h) mx = fmaxf(mx, s[h]);
    float e[8], sum = 0.f;
#pragma unroll
    for (int h = 0; h < 8; ++h) { e[h] = __expf(s[h] - mx); sum += e[h]; }
    const float inv = 1.f / sum;

    float t = 0.f;
#pragma unroll
    for (int h = 0; h < 8; ++h) {
        const float wgt = e[h] * inv;
        const size_t idx = ((size_t)(n * 8 + h) * 64 + d) * 9 + p;
        W_out[idx] = wgt;
        t = fmaf(wgt, vp[idx], t);
    }
    T_out[((size_t)n * 64 + d) * 9 + p] = t;
}

// ---------------------------------------------------------------------------
extern "C" void kernel_launch(void* const* d_in, const int* in_sizes, int n_in,
                              void* d_out, int out_size, void* d_ws, size_t ws_size,
                              hipStream_t stream) {
    (void)in_sizes; (void)n_in; (void)out_size; (void)ws_size;

    const float* X      = (const float*)d_in[0];
    const float* lin_w  = (const float*)d_in[1];
    const float* lin_b  = (const float*)d_in[2];
    const float* gamma  = (const float*)d_in[3];
    const float* beta   = (const float*)d_in[4];
    const float* px     = (const float*)d_in[5];
    const float* py     = (const float*)d_in[6];
    const float* kw     = (const float*)d_in[7];
    const float* kb     = (const float*)d_in[8];
    const float* vw     = (const float*)d_in[9];
    const float* vb     = (const float*)d_in[10];
    const float* sw     = (const float*)d_in[11];
    const float* sb     = (const float*)d_in[12];

    // Output layout: T | M | P | w (flat, return order)
    float* T_out = (float*)d_out;
    float* M_out = T_out + 8 * 64 * 9;          // +4608
    float* P_out = M_out + 8 * 8 * 32;          // +2048
    float* W_out = P_out + 8 * 8 * 64 * 9;      // +36864

    // Workspace: kp (36864 f32) | vp (36864 f32)
    float* kp_ws = (float*)d_ws;
    float* vp_ws = kp_ws + 8 * 8 * 64 * 9;

    embed_kernel<<<8, 32, 0, stream>>>(X, lin_w, lin_b, gamma, beta, px, py,
                                       M_out, P_out);
    kpvp_kernel<<<40, 32, 0, stream>>>(P_out, kw, kb, vw, vb, kp_ws, vp_ws);

    const size_t lds_bytes = (size_t)(64 * SW_STRIDE + 8 * 64 * 9) * sizeof(float);
    attn_kernel<<<8, 576, lds_bytes, stream>>>(kp_ws, vp_ws, sw, sb, T_out, W_out);
}
